// PaiNN_33225867002062
// MI455X (gfx1250) — matrix operand order repaired and checked by reading, not compiled, not run on hardware
//
// PaiNN forward for MI455X (gfx1250, wave32, WMMA).
//
// Strategy: factor the radial basis out of the message einsums so every heavy
// step is a dense GEMM executed with v_wmma_f32_16x16x32_bf16 (bf16 inputs,
// f32 accumulate -- LayerNorm after every block makes bf16 operands safe).
// The (N,N,3F) message tensor (472 MB) is never materialized; instead three
// GEMMs with K=N=320 produce per-(atom, rbf-row) partials P/Q/R that are
// contracted with sf_W in a cheap VALU kernel. Total matrix work ~35 GFLOP vs
// ~150 MB of workspace traffic -> compute-side, exactly what WMMA is for.
//
// GEMM data path (CDNA5-tuned): 4-wave workgroup owns a 16x128 C tile; each
// 32-wide K-slab of A and B is staged cooperatively into LDS with coalesced
// float4 (global_load_b128) reads, converted to bf16 once, B transposed to
// [n][k] so every lane fragment is a contiguous 32B LDS read (ds_load_b128).
// Each wave reuses its A fragment across two 16-col WMMA subtiles.

#include <hip/hip_runtime.h>
#include <hip/hip_bf16.h>
#include <math.h>

#define NA   320          // atoms
#define FD   384          // feature dim
#define FD3  1152         // 3*FD
#define NRBF 20
#define RPAD 32           // rbf rows padded for M/K alignment
#define EPAD 64           // 21*3 = 63 rbf*dir rows padded
#define NGRAPH 32
#define APG  10           // atoms per graph
#define CUTOFF_F 5.0f
#define PI_F 3.14159265358979323846f

typedef __attribute__((ext_vector_type(16))) __bf16 bf16x16;
typedef __attribute__((ext_vector_type(8)))  __bf16 bf16x8;
typedef __attribute__((ext_vector_type(8)))  float  f32x8;

// ---------------------------------------------------------------- reductions
__device__ __forceinline__ float block_reduce_sum(float x, float* red) {
    const int t = threadIdx.x;
    red[t] = x;
    __syncthreads();
    for (int s = blockDim.x >> 1; s > 0; s >>= 1) {
        if (t < s) red[t] += red[t + s];
        __syncthreads();
    }
    const float r = red[0];
    __syncthreads();
    return r;
}

__device__ __forceinline__ float silu_f(float x) {
    return x / (1.0f + __expf(-x));
}

// ------------------------------------------------------------ generic WMMA GEMM
// C[M,N] = act(A[M,K] @ B + bias). BT=1 reads B as [N,K] (transposed weights).
// Requires M%16==0, K%32==0, N%128==0. Block = 128 thr (4 waves) -> 16x128 C.
// Fragments: A 16x32, B 32x16 with contiguous k (lane half h holds k 16h..16h+15), f32 C 16x16.
template <int BT, int SILU, int BIAS>
__global__ void k_gemm(const float* __restrict__ A, const float* __restrict__ Bm,
                       const float* __restrict__ bias, float* __restrict__ C,
                       int M, int K, int Nn) {
    __shared__ __bf16 ldsA[16 * 32];        // [m][k]
    __shared__ __bf16 ldsB[128 * 32];       // [n][k] (transposed for contiguous frags)

    const int lane = threadIdx.x & 31;
    const int wid  = threadIdx.x >> 5;
    const int tilesN = Nn >> 7;             // 128-wide block tiles
    const int m0  = (blockIdx.x / tilesN) << 4;
    const int nn0 = (blockIdx.x % tilesN) << 7;
    const int l15 = lane & 15;
    const bool hi = lane >= 16;
    const int kbA = hi ? 16 : 0;            // A: lanes16-31 hold K 16..31
    const int kbB = hi ? 16 : 0;            // B: lanes16-31 hold K 16..31

    f32x8 acc0 = {};
    f32x8 acc1 = {};

    for (int k0 = 0; k0 < K; k0 += 32) {
        __syncthreads();                    // protect previous iteration's reads
        // ---- stage A slab 16x32 (1 float4 per thread, coalesced) ----
        {
            const int idx = threadIdx.x << 2;   // 0..508
            const int row = idx >> 5;
            const int col = idx & 31;
            const float4 a4 =
                *(const float4*)(A + (size_t)(m0 + row) * K + k0 + col);
            __bf16* d = &ldsA[row * 32 + col];
            d[0] = (__bf16)a4.x; d[1] = (__bf16)a4.y;
            d[2] = (__bf16)a4.z; d[3] = (__bf16)a4.w;
            if (k0 + 32 < K)
                __builtin_prefetch(A + (size_t)(m0 + row) * K + k0 + 32 + col, 0, 1);
        }
        // ---- stage B slab 32x128 (8 float4 per thread, coalesced) ----
#pragma unroll
        for (int q = 0; q < 8; ++q) {
            const int idx = (threadIdx.x + (q << 7)) << 2;  // 0..4092
            if (BT) {
                const int nc = idx >> 5;
                const int kr = idx & 31;
                const float4 b4 =
                    *(const float4*)(Bm + (size_t)(nn0 + nc) * K + k0 + kr);
                __bf16* d = &ldsB[nc * 32 + kr];
                d[0] = (__bf16)b4.x; d[1] = (__bf16)b4.y;
                d[2] = (__bf16)b4.z; d[3] = (__bf16)b4.w;
            } else {
                const int kr = idx >> 7;
                const int nc = idx & 127;
                const float4 b4 =
                    *(const float4*)(Bm + (size_t)(k0 + kr) * Nn + nn0 + nc);
                ldsB[(nc + 0) * 32 + kr] = (__bf16)b4.x;
                ldsB[(nc + 1) * 32 + kr] = (__bf16)b4.y;
                ldsB[(nc + 2) * 32 + kr] = (__bf16)b4.z;
                ldsB[(nc + 3) * 32 + kr] = (__bf16)b4.w;
            }
        }
        __syncthreads();

        // ---- fragments from LDS (ds_load_b128) + 2 WMMAs (A reused) ----
        bf16x16 af;
        {
            const bf16x8 alo = *(const bf16x8*)&ldsA[l15 * 32 + kbA];
            const bf16x8 ahi = *(const bf16x8*)&ldsA[l15 * 32 + kbA + 8];
#pragma unroll
            for (int e = 0; e < 8; ++e) { af[e] = alo[e]; af[e + 8] = ahi[e]; }
        }
        const bf16x16 bf0 =
            *(const bf16x16*)&ldsB[(wid * 32 + l15) * 32 + kbB];
        const bf16x16 bf1 =
            *(const bf16x16*)&ldsB[(wid * 32 + 16 + l15) * 32 + kbB];
        acc0 = __builtin_amdgcn_wmma_f32_16x16x32_bf16(
            false, af, false, bf0, (short)0, acc0, false, false);
        acc1 = __builtin_amdgcn_wmma_f32_16x16x32_bf16(
            false, af, false, bf1, (short)0, acc1, false, false);
    }

    // ---- epilogue ----
#pragma unroll
    for (int j = 0; j < 2; ++j) {
        const f32x8 acc = j ? acc1 : acc0;
        const int ncol = nn0 + wid * 32 + j * 16 + l15;
        const float bn = BIAS ? bias[ncol] : 0.0f;
#pragma unroll
        for (int r = 0; r < 8; ++r) {
            const int m = m0 + r + (hi ? 8 : 0);
            float x = acc[r] + bn;
            if (SILU) x = silu_f(x);
            C[(size_t)m * Nn + ncol] = x;
        }
    }
}

// --------------------------------------------------- geometry + rbf factor rows
// rbA[(i*32+r)*NA + j] = r<20: sin(d*pre_r)*fcut ; r==20: fcut ; else 0
// rbE[(i*64+(r*3+v))*NA + j] = rbA_row(r) * e_ij[v]   (63 rows used, padded to 64)
__global__ void k_geom(const float* __restrict__ pos,
                       float* __restrict__ rbA, float* __restrict__ rbE) {
    __builtin_amdgcn_s_wait_tensorcnt(0);
    const int t = blockIdx.x * blockDim.x + threadIdx.x;
    if (t >= NA * NA) return;
    const int i = t / NA, j = t % NA;
    const float dx = pos[j * 3 + 0] - pos[i * 3 + 0];
    const float dy = pos[j * 3 + 1] - pos[i * 3 + 1];
    const float dz = pos[j * 3 + 2] - pos[i * 3 + 2];
    const float d  = sqrtf(dx * dx + dy * dy + dz * dz + 1e-12f);
    const float fc = (d < CUTOFF_F) ? 0.5f * (__cosf(PI_F * d / CUTOFF_F) + 1.0f) : 0.0f;
    const float inv = 1.0f / (d + 1e-9f);
    const float ev[3] = {dx * inv, dy * inv, dz * inv};

    float rb[NRBF + 1];
#pragma unroll
    for (int r = 0; r < NRBF; ++r)
        rb[r] = __sinf(d * ((float)(r + 1) * PI_F / CUTOFF_F)) * fc;
    rb[NRBF] = fc;  // carries sf_b * fcut

#pragma unroll
    for (int r = 0; r < RPAD; ++r)
        rbA[((size_t)i * RPAD + r) * NA + j] = (r <= NRBF) ? rb[r] : 0.0f;
#pragma unroll
    for (int r = 0; r < EPAD; ++r) {
        float val = 0.0f;
        if (r < 63) val = rb[r / 3] * ev[r % 3];
        rbE[((size_t)i * EPAD + r) * NA + j] = val;
    }
}

// ------------------------------------------ masked phi right-hand-side matrices
__global__ void k_phim(const float* __restrict__ phi, const float* __restrict__ v,
                       const float* __restrict__ mask,
                       float* __restrict__ pss, float* __restrict__ p2,
                       float* __restrict__ pvs) {
    const int t = blockIdx.x * blockDim.x + threadIdx.x;
    if (t >= NA * FD) return;
    const int j = t / FD, f = t % FD;
    const float mk = mask[j];
    pss[t] = phi[(size_t)j * FD3 + f] * mk;
    const float pv = phi[(size_t)j * FD3 + FD + f] * mk;
    p2[(size_t)j * FD3 + f * 3 + 0] = v[((size_t)j * FD + f) * 3 + 0] * pv;
    p2[(size_t)j * FD3 + f * 3 + 1] = v[((size_t)j * FD + f) * 3 + 1] * pv;
    p2[(size_t)j * FD3 + f * 3 + 2] = v[((size_t)j * FD + f) * 3 + 2] * pv;
    pvs[t] = phi[(size_t)j * FD3 + 2 * FD + f] * mk;
}

// ---------------------------- contract P/Q/R with sf_W' -> ds,dv; add residual
__global__ void k_msg_reduce(const float* __restrict__ P, const float* __restrict__ Q,
                             const float* __restrict__ R,
                             const float* __restrict__ sfW, const float* __restrict__ sfb,
                             const float* __restrict__ s, const float* __restrict__ v,
                             float* __restrict__ sPre, float* __restrict__ vPre) {
    const int t = blockIdx.x * blockDim.x + threadIdx.x;
    if (t >= NA * FD) return;
    const int i = t / FD, f = t % FD;
    float ds = 0.0f, dv0 = 0.0f, dv1 = 0.0f, dv2 = 0.0f;
    for (int r = 0; r <= NRBF; ++r) {
        const float wss = (r < NRBF) ? sfW[(size_t)r * FD3 + f] : sfb[f];
        const float wvv = (r < NRBF) ? sfW[(size_t)r * FD3 + FD + f] : sfb[FD + f];
        const float wvs = (r < NRBF) ? sfW[(size_t)r * FD3 + 2 * FD + f] : sfb[2 * FD + f];
        ds  += P[((size_t)i * RPAD + r) * FD + f] * wss;
        dv0 += Q[((size_t)i * RPAD + r) * FD3 + f * 3 + 0] * wvv;
        dv1 += Q[((size_t)i * RPAD + r) * FD3 + f * 3 + 1] * wvv;
        dv2 += Q[((size_t)i * RPAD + r) * FD3 + f * 3 + 2] * wvv;
        dv0 += R[((size_t)i * EPAD + r * 3 + 0) * FD + f] * wvs;
        dv1 += R[((size_t)i * EPAD + r * 3 + 1) * FD + f] * wvs;
        dv2 += R[((size_t)i * EPAD + r * 3 + 2) * FD + f] * wvs;
    }
    sPre[t] = s[t] + ds;
    vPre[(size_t)t * 3 + 0] = v[(size_t)t * 3 + 0] + dv0;
    vPre[(size_t)t * 3 + 1] = v[(size_t)t * 3 + 1] + dv1;
    vPre[(size_t)t * 3 + 2] = v[(size_t)t * 3 + 2] + dv2;
}

// ----------------------------------------------------------- LayerNorm over FD
__global__ void k_ln(const float* __restrict__ x, const float* __restrict__ g,
                     const float* __restrict__ b, float* __restrict__ y) {
    __shared__ float red[128];
    const int i = blockIdx.x, t = threadIdx.x;
    float s0 = 0.0f;
    for (int f = t; f < FD; f += 128) s0 += x[(size_t)i * FD + f];
    const float mean = block_reduce_sum(s0, red) * (1.0f / FD);
    float s1 = 0.0f;
    for (int f = t; f < FD; f += 128) {
        const float d = x[(size_t)i * FD + f] - mean;
        s1 += d * d;
    }
    const float var = block_reduce_sum(s1, red) * (1.0f / FD);
    const float inv = rsqrtf(var + 1e-5f);
    for (int f = t; f < FD; f += 128) {
        const float gg = g ? g[f] : 1.0f;
        const float bb = b ? b[f] : 0.0f;
        y[(size_t)i * FD + f] = gg * (x[(size_t)i * FD + f] - mean) * inv + bb;
    }
}

// ------------------------------------------- vector renorm (LN of channel norms)
__global__ void k_vren(const float* __restrict__ vin, const float* __restrict__ g,
                       const float* __restrict__ b, float* __restrict__ vout) {
    __shared__ float vn[FD];
    __shared__ float red[128];
    const int i = blockIdx.x, t = threadIdx.x;
    for (int f = t; f < FD; f += 128) {
        const float* p = vin + ((size_t)i * FD + f) * 3;
        vn[f] = sqrtf(p[0] * p[0] + p[1] * p[1] + p[2] * p[2] + 1e-12f);
    }
    __syncthreads();
    float s0 = 0.0f;
    for (int f = t; f < FD; f += 128) s0 += vn[f];
    const float mean = block_reduce_sum(s0, red) * (1.0f / FD);
    float s1 = 0.0f;
    for (int f = t; f < FD; f += 128) {
        const float d = vn[f] - mean;
        s1 += d * d;
    }
    const float var = block_reduce_sum(s1, red) * (1.0f / FD);
    const float inv = rsqrtf(var + 1e-5f);
    for (int f = t; f < FD; f += 128) {
        const float gg = g ? g[f] : 1.0f;
        const float bb = b ? b[f] : 0.0f;
        const float ln = gg * (vn[f] - mean) * inv + bb;
        const float sc = ln / (vn[f] + 1e-9f);
        const float* p = vin + ((size_t)i * FD + f) * 3;
        float* q = vout + ((size_t)i * FD + f) * 3;
        q[0] = p[0] * sc; q[1] = p[1] * sc; q[2] = p[2] * sc;
    }
}

// ----------------------------------- repack v (N,F,3) -> X (N*3, F) for Wvv GEMM
__global__ void k_pack_v(const float* __restrict__ v, float* __restrict__ X) {
    const int t = blockIdx.x * blockDim.x + threadIdx.x;
    if (t >= NA * FD * 3) return;
    const int i = t / (FD * 3), rem = t % (FD * 3);
    const int f = rem / 3, vd = rem % 3;
    X[((size_t)i * 3 + vd) * FD + f] = v[((size_t)i * FD + f) * 3 + vd];
}

// ------------------- ||Vv||, <Vv,Uv>, and the [s, ||Vv||] concat for update MLP
__global__ void k_catnorm(const float* __restrict__ Vv, const float* __restrict__ Uv,
                          const float* __restrict__ s,
                          float* __restrict__ cat, float* __restrict__ spb) {
    const int t = blockIdx.x * blockDim.x + threadIdx.x;
    if (t >= NA * FD) return;
    const int i = t / FD, gg = t % FD;
    float n2 = 1e-12f, sp = 0.0f;
#pragma unroll
    for (int vd = 0; vd < 3; ++vd) {
        const float a = Vv[((size_t)i * 3 + vd) * FD + gg];
        const float u = Uv[((size_t)i * 3 + vd) * FD + gg];
        n2 += a * a;
        sp += a * u;
    }
    cat[(size_t)i * (2 * FD) + gg]      = s[t];
    cat[(size_t)i * (2 * FD) + FD + gg] = sqrtf(n2);
    spb[t] = sp;
}

// ------------------------------------------------------- update-block residuals
__global__ void k_upd(const float* __restrict__ a, const float* __restrict__ sp,
                      const float* __restrict__ Uv,
                      const float* __restrict__ s, const float* __restrict__ v,
                      float* __restrict__ sPre, float* __restrict__ vPre) {
    const int t = blockIdx.x * blockDim.x + threadIdx.x;
    if (t >= NA * FD) return;
    const int i = t / FD, f = t % FD;
    const float a0 = a[(size_t)i * FD3 + f];
    const float a1 = a[(size_t)i * FD3 + FD + f];
    const float a2 = a[(size_t)i * FD3 + 2 * FD + f];
    sPre[t] = s[t] + a0 + a1 * sp[t];
#pragma unroll
    for (int vd = 0; vd < 3; ++vd)
        vPre[(size_t)t * 3 + vd] =
            v[(size_t)t * 3 + vd] + a2 * Uv[((size_t)i * 3 + vd) * FD + f];
}

// -------------------------------------------------------------- readout energy
__global__ void k_energy(const float* __restrict__ pre, const float* __restrict__ rW2,
                         const float* __restrict__ rb2, const float* __restrict__ s,
                         float* __restrict__ out) {
    __shared__ float red[128];
    const int g = blockIdx.x, t = threadIdx.x;
    float E = 0.0f;
    for (int a = 0; a < APG; ++a) {
        const int i = g * APG + a;
        float pr = 0.0f, ss = 0.0f;
        for (int f = t; f < FD; f += 128) {
            pr += pre[(size_t)i * FD + f] * rW2[f];
            ss += s[(size_t)i * FD + f];
        }
        const float ro   = block_reduce_sum(pr, red) + rb2[0];
        const float ssum = block_reduce_sum(ss, red);
        E += ro * ssum;
    }
    if (t == 0) out[g] = E;
}

__global__ void k_copy(const float* __restrict__ src, float* __restrict__ dst, int n) {
    const int t = blockIdx.x * blockDim.x + threadIdx.x;
    if (t < n) dst[t] = src[t];
}

// ------------------------------------------------------------------ workspace
#define NF   ((size_t)NA * FD)        // 122880
#define NF3L ((size_t)NA * FD3)       // 368640
#define NV3  ((size_t)NA * FD * 3)    // 368640
static constexpr size_t OFF_S   = 0;
static constexpr size_t OFF_V   = OFF_S   + NF;
static constexpr size_t OFF_SP  = OFF_V   + NV3;
static constexpr size_t OFF_VP  = OFF_SP  + NF;
static constexpr size_t OFF_H   = OFF_VP  + NV3;
static constexpr size_t OFF_PHI = OFF_H   + NF;        // phi, later 'a'
static constexpr size_t OFF_PSS = OFF_PHI + NF3L;
static constexpr size_t OFF_P2  = OFF_PSS + NF;
static constexpr size_t OFF_PVS = OFF_P2  + NF3L;
static constexpr size_t OFF_RBA = OFF_PVS + NF;
static constexpr size_t OFF_RBE = OFF_RBA + (size_t)NA * RPAD * NA;
static constexpr size_t OFF_P   = OFF_RBE + (size_t)NA * EPAD * NA;
static constexpr size_t OFF_Q   = OFF_P   + (size_t)NA * RPAD * FD;
static constexpr size_t OFF_R   = OFF_Q   + (size_t)NA * RPAD * FD3;
static constexpr size_t OFF_X   = OFF_R   + (size_t)NA * EPAD * FD;
static constexpr size_t OFF_VV  = OFF_X   + NV3;
static constexpr size_t OFF_UV  = OFF_VV  + NV3;
static constexpr size_t OFF_SPR = OFF_UV  + NV3;
static constexpr size_t OFF_CAT = OFF_SPR + NF;

static void gemm_launch(const float* A, const float* B, const float* bias, float* C,
                        int M, int K, int Nn, bool bt, bool act, bool hasb,
                        hipStream_t st) {
    const int blocks = (M / 16) * (Nn / 128);  // exact grid: EXEC all-1s for WMMA
    if (bt)            k_gemm<1, 0, 0><<<blocks, 128, 0, st>>>(A, B, bias, C, M, K, Nn);
    else if (act)      k_gemm<0, 1, 1><<<blocks, 128, 0, st>>>(A, B, bias, C, M, K, Nn);
    else if (hasb)     k_gemm<0, 0, 1><<<blocks, 128, 0, st>>>(A, B, bias, C, M, K, Nn);
    else               k_gemm<0, 0, 0><<<blocks, 128, 0, st>>>(A, B, bias, C, M, K, Nn);
}

extern "C" void kernel_launch(void* const* d_in, const int* in_sizes, int n_in,
                              void* d_out, int out_size, void* d_ws, size_t ws_size,
                              hipStream_t stream) {
    (void)in_sizes; (void)n_in; (void)out_size; (void)ws_size;
    // jax pytree flatten order (dict keys sorted):
    // 0 s, 1 v, 2 pos, 3 atom_mask, 4 distances, 5 directions,
    // 6 emb_W, 7 emb_b,
    // layers[0..1] each: Wuv, Wvv, lns_b, lns_g, lnv_b, lnv_g,
    //                    mlp_W1, mlp_W2, mlp_b1, mlp_b2,
    //                    phi_W1, phi_W2, phi_b1, phi_b2, sf_W, sf_b   (16 each)
    // 40 out_W, 41 out_b, 42 r_W1, 43 r_b1, 44 r_W2, 45 r_b2
    const float* in_s  = (const float*)d_in[0];
    const float* in_v  = (const float*)d_in[1];
    const float* pos   = (const float*)d_in[2];
    const float* mask  = (const float*)d_in[3];
    const float* embW  = (const float*)d_in[6];
    const float* embB  = (const float*)d_in[7];
    const float* outW  = (const float*)d_in[40];
    const float* outB  = (const float*)d_in[41];
    const float* rW1   = (const float*)d_in[42];
    const float* rb1   = (const float*)d_in[43];
    const float* rW2   = (const float*)d_in[44];
    const float* rb2   = (const float*)d_in[45];

    float* W = (float*)d_ws;
    float* s    = W + OFF_S;
    float* v    = W + OFF_V;
    float* sPre = W + OFF_SP;
    float* vPre = W + OFF_VP;
    float* h    = W + OFF_H;
    float* phi  = W + OFF_PHI;   // also 'a'
    float* pss  = W + OFF_PSS;
    float* p2   = W + OFF_P2;
    float* pvs  = W + OFF_PVS;
    float* rbA  = W + OFF_RBA;
    float* rbE  = W + OFF_RBE;
    float* Pm   = W + OFF_P;
    float* Qm   = W + OFF_Q;
    float* Rm   = W + OFF_R;
    float* Xv   = W + OFF_X;
    float* Vv   = W + OFF_VV;
    float* Uv   = W + OFF_UV;
    float* spb  = W + OFF_SPR;
    float* cat  = W + OFF_CAT;

    const int TPB = 256;
    const int nfBlocks  = (NA * FD + TPB - 1) / TPB;
    const int nv3Blocks = (NA * FD * 3 + TPB - 1) / TPB;

    // geometry -> factored rbf rows (shared by both layers; mask folded into phim)
    k_geom<<<(NA * NA + TPB - 1) / TPB, TPB, 0, stream>>>(pos, rbA, rbE);

    // embedding + initial vector renorm (unit affine)
    gemm_launch(in_s, embW, embB, s, NA, FD, FD, false, false, true, stream);
    k_vren<<<NA, 128, 0, stream>>>(in_v, nullptr, nullptr, v);

    for (int L = 0; L < 2; ++L) {
        const int base = 8 + L * 16;
        const float* Wuv  = (const float*)d_in[base + 0];
        const float* Wvv  = (const float*)d_in[base + 1];
        const float* lnsB = (const float*)d_in[base + 2];
        const float* lnsG = (const float*)d_in[base + 3];
        const float* lnvB = (const float*)d_in[base + 4];
        const float* lnvG = (const float*)d_in[base + 5];
        const float* mW1  = (const float*)d_in[base + 6];
        const float* mW2  = (const float*)d_in[base + 7];
        const float* mb1  = (const float*)d_in[base + 8];
        const float* mb2  = (const float*)d_in[base + 9];
        const float* pW1  = (const float*)d_in[base + 10];
        const float* pW2  = (const float*)d_in[base + 11];
        const float* pb1  = (const float*)d_in[base + 12];
        const float* pb2  = (const float*)d_in[base + 13];
        const float* sfW  = (const float*)d_in[base + 14];
        const float* sfb  = (const float*)d_in[base + 15];

        // ---- MessageBlock ----
        gemm_launch(s, pW1, pb1, h, NA, FD, FD, false, true, true, stream);      // silu
        gemm_launch(h, pW2, pb2, phi, NA, FD, FD3, false, false, true, stream);
        k_phim<<<nfBlocks, TPB, 0, stream>>>(phi, v, mask, pss, p2, pvs);
        gemm_launch(rbA, pss, nullptr, Pm, NA * RPAD, NA, FD,  false, false, false, stream);
        gemm_launch(rbA, p2,  nullptr, Qm, NA * RPAD, NA, FD3, false, false, false, stream);
        gemm_launch(rbE, pvs, nullptr, Rm, NA * EPAD, NA, FD,  false, false, false, stream);
        k_msg_reduce<<<nfBlocks, TPB, 0, stream>>>(Pm, Qm, Rm, sfW, sfb, s, v, sPre, vPre);
        k_ln<<<NA, 128, 0, stream>>>(sPre, lnsG, lnsB, s);
        k_vren<<<NA, 128, 0, stream>>>(vPre, lnvG, lnvB, v);

        // ---- UpdateBlock ----
        k_pack_v<<<nv3Blocks, TPB, 0, stream>>>(v, Xv);
        gemm_launch(Xv, Wvv, nullptr, Vv, NA * 3, FD, FD, true, false, false, stream);
        gemm_launch(Xv, Wuv, nullptr, Uv, NA * 3, FD, FD, true, false, false, stream);
        k_catnorm<<<nfBlocks, TPB, 0, stream>>>(Vv, Uv, s, cat, spb);
        gemm_launch(cat, mW1, mb1, h, NA, 2 * FD, FD, false, true, true, stream);  // silu
        gemm_launch(h, mW2, mb2, phi, NA, FD, FD3, false, false, true, stream);    // 'a'
        k_upd<<<nfBlocks, TPB, 0, stream>>>(phi, spb, Uv, s, v, sPre, vPre);
        k_ln<<<NA, 128, 0, stream>>>(sPre, lnsG, lnsB, s);
        k_vren<<<NA, 128, 0, stream>>>(vPre, lnvG, lnvB, v);
    }

    // ---- Readout ----
    float* out = (float*)d_out;
    gemm_launch(s, rW1, rb1, h, NA, FD, FD, false, true, true, stream);           // silu
    k_energy<<<NGRAPH, 128, 0, stream>>>(h, rW2, rb2, s, out);                    // [0,32)
    gemm_launch(s, outW, outB, out + NGRAPH, NA, FD, 128, false, false, true, stream);
    k_copy<<<nv3Blocks, TPB, 0, stream>>>(v, out + NGRAPH + NA * 128, NA * FD * 3);
}
